// DynNetworkold_35562329211365
// MI455X (gfx1250) — matrix operand order repaired and checked by reading, not compiled, not run on hardware
//
#include <hip/hip_runtime.h>
#include <hip/hip_bf16.h>

typedef _Float16 half8  __attribute__((ext_vector_type(8)));
typedef _Float16 half16 __attribute__((ext_vector_type(16)));
typedef float    floatx8 __attribute__((ext_vector_type(8)));

#define SIM_TIME 100
#define BETA     0.9f
#define B_       256
#define D_IN     700
#define D_IN_P   704      // padded to multiple of 32 (22 k-blocks)
#define H_       1024
#define D_OUT    20
#define D_OUT_P  32
#define NBLOCKS  64       // persistent grid: 64 blocks x 128 thr = 256 waves

union Frag { half16 v; half8 h[2]; };

// ---- fragment loaders (wave32, v_wmma_f32_16x16x32_f16 layouts) ----
// A (MxK row-major): lane = row m0+(lane&15); element e -> K = kb*32 + (lane>>4)*8 + e%8 + (e/8)*16
__device__ inline half16 load_a_frag(const _Float16* S, int pitch, int m0, int kb, int lane) {
    int row = m0 + (lane & 15);
    int k0  = kb * 32 + ((lane >> 4) << 3);
    const _Float16* p = S + row * pitch + k0;
    Frag f;
    f.h[0] = *(const half8*)(p);
    f.h[1] = *(const half8*)(p + 16);
    return f.v;
}
// B (KxN) from weights stored [N][K] row-major: lane = col n0+(lane&15);
// element e -> K = kb*32 + (lane>>4)*8 + e%8 + (e/8)*16
__device__ inline half16 load_b_frag(const _Float16* W, int pitch, int n0, int kb, int lane) {
    int row = n0 + (lane & 15);
    int k0  = kb * 32 + ((lane >> 4) << 3);
    const _Float16* p = W + row * pitch + k0;
    Frag f;
    f.h[0] = *(const half8*)(p);
    f.h[1] = *(const half8*)(p + 16);
    return f.v;
}

template<int NT>
__device__ inline void wmma_rowtile(const _Float16* A, int lda,
                                    const _Float16* W, int ldw,
                                    int m0, int n0, int kblocks, int lane,
                                    floatx8* acc) {
    floatx8 z = {};
#pragma unroll
    for (int t = 0; t < NT; ++t) acc[t] = z;
    for (int kb = 0; kb < kblocks; ++kb) {
        half16 a = load_a_frag(A, lda, m0, kb, lane);
#pragma unroll
        for (int t = 0; t < NT; ++t) {
            half16 b = load_b_frag(W, ldw, n0 + 16 * t, kb, lane);
            acc[t] = __builtin_amdgcn_wmma_f32_16x16x32_f16(
                false, a, false, b, (short)0, acc[t], false, false);
        }
    }
}

// ---- sense-reversing grid barrier (persistent kernel) ----
__device__ inline void grid_barrier(int* count, int* gen) {
    __syncthreads();
    if (threadIdx.x == 0) {
        __threadfence();   // release: make this block's stores device-visible
        int g = __hip_atomic_load(gen, __ATOMIC_RELAXED, __HIP_MEMORY_SCOPE_AGENT);
        if (__hip_atomic_fetch_add(count, 1, __ATOMIC_ACQ_REL, __HIP_MEMORY_SCOPE_AGENT)
            == NBLOCKS - 1) {
            __hip_atomic_store(count, 0, __ATOMIC_RELAXED, __HIP_MEMORY_SCOPE_AGENT);
            __hip_atomic_fetch_add(gen, 1, __ATOMIC_ACQ_REL, __HIP_MEMORY_SCOPE_AGENT);
        } else {
            while (__hip_atomic_load(gen, __ATOMIC_ACQUIRE, __HIP_MEMORY_SCOPE_AGENT) == g)
                __builtin_amdgcn_s_sleep(8);
        }
        __threadfence();   // acquire: invalidate stale lines before next phase
    }
    __syncthreads();
}

// ---------------- prep kernels ----------------
__global__ void slice_pad_f16_kernel(const float* __restrict__ src, _Float16* __restrict__ dst,
                                     int dst_rows, int src_rows, int src_pitch,
                                     int col0, int copy_cols, int dst_pitch) {
    int i = blockIdx.x * blockDim.x + threadIdx.x;
    if (i >= dst_rows * dst_pitch) return;
    int r = i / dst_pitch, c = i - r * dst_pitch;
    float v = 0.f;
    if (r < src_rows && c < copy_cols) v = src[r * src_pitch + col0 + c];
    dst[i] = (_Float16)v;
}

__global__ void init_state_kernel(const float* __restrict__ s1_0, const float* __restrict__ m1_0,
                                  const float* __restrict__ m2_0, const float* __restrict__ mo_0,
                                  const float* __restrict__ mm_0,
                                  _Float16* __restrict__ s1, float* __restrict__ m1,
                                  float* __restrict__ m2, float* __restrict__ mo,
                                  float* __restrict__ mm, int* __restrict__ bar) {
    int i = blockIdx.x * blockDim.x + threadIdx.x;
    if (i < B_ * H_) {
        s1[i] = (_Float16)s1_0[i];
        m1[i] = m1_0[i];
        m2[i] = m2_0[i];
    }
    if (i < B_ * D_OUT_P) {
        int b = i / D_OUT_P, c = i - b * D_OUT_P;
        float vo = 0.f, vm = 0.f;
        if (c < D_OUT) { vo = mo_0[b * D_OUT + c]; vm = mm_0[b * D_OUT + c]; }
        mo[i] = vo; mm[i] = vm;
    }
    if (i < 2) bar[i * 64] = 0;   // count @0, gen @64 (separate lines)
}

// loop-invariant input current: cur1 = inp @ W1in^T + b1
__global__ void base_kernel(const _Float16* __restrict__ inp16, const _Float16* __restrict__ W1in,
                            const float* __restrict__ b1, float* __restrict__ cur1) {
    int lane = threadIdx.x & 31;
    int tid  = blockIdx.x * 4 + (threadIdx.x >> 5);
    int m0 = (tid >> 4) * 16;
    int n0 = (tid & 15) * 64;
    floatx8 acc[4];
    wmma_rowtile<4>(inp16, D_IN_P, W1in, D_IN_P, m0, n0, D_IN_P / 32, lane, acc);
#pragma unroll
    for (int t = 0; t < 4; ++t) {
        int h = n0 + t * 16 + (lane & 15);
        float bias = b1[h];
#pragma unroll
        for (int r = 0; r < 8; ++r) {
            int b = m0 + r + ((lane >> 4) << 3);
            cur1[b * H_ + h] = acc[t][r] + bias;
        }
    }
}

// ---------------- per-phase device bodies ----------------
__device__ inline void layer1_body(int wtid, int lane, const _Float16* s1p,
                                   const float* cur1, float* m1,
                                   const _Float16* W1rec, _Float16* s1n) {
    int m0 = (wtid >> 4) * 16;
    int n0 = (wtid & 15) * 64;
    floatx8 acc[4];
    wmma_rowtile<4>(s1p, H_, W1rec, H_, m0, n0, H_ / 32, lane, acc);
#pragma unroll
    for (int t = 0; t < 4; ++t) {
        int h = n0 + t * 16 + (lane & 15);
#pragma unroll
        for (int r = 0; r < 8; ++r) {
            int b = m0 + r + ((lane >> 4) << 3);
            int idx = b * H_ + h;
            float nv  = BETA * m1[idx] + acc[t][r] + cur1[idx];
            float spk = (nv >= 1.0f) ? 1.0f : 0.0f;
            m1[idx] = nv * (1.0f - spk);
            s1n[idx] = (_Float16)spk;
        }
    }
}

__device__ inline void layer2_body(int wtid, int lane, const _Float16* s1n,
                                   const float* b2, float* m2,
                                   const _Float16* W2, _Float16* s2) {
    int m0 = (wtid >> 4) * 16;
    int n0 = (wtid & 15) * 64;
    floatx8 acc[4];
    wmma_rowtile<4>(s1n, H_, W2, H_, m0, n0, H_ / 32, lane, acc);
#pragma unroll
    for (int t = 0; t < 4; ++t) {
        int h = n0 + t * 16 + (lane & 15);
        float bias = b2[h];
#pragma unroll
        for (int r = 0; r < 8; ++r) {
            int b = m0 + r + ((lane >> 4) << 3);
            int idx = b * H_ + h;
            float nv  = BETA * m2[idx] + acc[t][r] + bias;
            float spk = (nv >= 1.0f) ? 1.0f : 0.0f;
            m2[idx] = nv * (1.0f - spk);
            s2[idx] = (_Float16)spk;
        }
    }
}

__device__ inline void head_body(int wtid, int lane, const _Float16* s2,
                                 float* mo, float* mm,
                                 const _Float16* Wo16, const _Float16* Wm16,
                                 const float* bo, const float* bm,
                                 float* out, int is_last) {
    int head = wtid >> 5;                 // 0: spiking out / mo, 1: LI mem / mm
    int m0 = ((wtid >> 1) & 15) * 16;
    int n0 = (wtid & 1) * 16;
    const _Float16* W = head ? Wm16 : Wo16;
    floatx8 acc[1];
    wmma_rowtile<1>(s2, H_, W, H_, m0, n0, H_ / 32, lane, acc);
    int n = n0 + (lane & 15);
    float bias = 0.f;
    if (n < D_OUT) bias = head ? bm[n] : bo[n];
    float* mem = head ? mm : mo;
#pragma unroll
    for (int r = 0; r < 8; ++r) {
        int b = m0 + r + ((lane >> 4) << 3);
        int idx = b * D_OUT_P + n;
        float nv = BETA * mem[idx] + acc[0][r] + bias;
        if (head == 0) {
            float spk = (nv >= 1.0f) ? 1.0f : 0.0f;
            mem[idx] = nv * (1.0f - spk);
            if (is_last && n < D_OUT) out[b * D_OUT + n] = spk;
        } else {
            mem[idx] = nv;
            if (is_last && n < D_OUT) out[B_ * D_OUT + b * D_OUT + n] = nv;
        }
    }
}

// ---------------- persistent timestep kernel: all 100 steps in one launch ----------------
__global__ void __launch_bounds__(128)
snn_persistent(const _Float16* __restrict__ W1rec, const _Float16* __restrict__ W2,
               const _Float16* __restrict__ Wo16, const _Float16* __restrict__ Wm16,
               const float* __restrict__ b2, const float* __restrict__ bo,
               const float* __restrict__ bm, const float* __restrict__ cur1,
               float* m1, float* m2, float* mo, float* mm,
               _Float16* s1a, _Float16* s1b, _Float16* s2,
               float* out, int* bar) {
    int lane = threadIdx.x & 31;
    int wtid = blockIdx.x * 4 + (threadIdx.x >> 5);   // 0..255
    int* count = bar;
    int* gen   = bar + 64;

    _Float16* scur = s1a;
    _Float16* snxt = s1b;
#pragma unroll 1
    for (int t = 0; t < SIM_TIME; ++t) {
        layer1_body(wtid, lane, scur, cur1, m1, W1rec, snxt);
        grid_barrier(count, gen);
        layer2_body(wtid, lane, snxt, b2, m2, W2, s2);
        grid_barrier(count, gen);
        if (wtid < 64)   // wave-uniform: EXEC stays all-1s for WMMA
            head_body(wtid, lane, s2, mo, mm, Wo16, Wm16, bo, bm, out,
                      (t == SIM_TIME - 1) ? 1 : 0);
        grid_barrier(count, gen);
        _Float16* tmp = scur; scur = snxt; snxt = tmp;
    }
}

extern "C" void kernel_launch(void* const* d_in, const int* in_sizes, int n_in,
                              void* d_out, int out_size, void* d_ws, size_t ws_size,
                              hipStream_t stream) {
    const float* inp  = (const float*)d_in[0];
    const float* s1_0 = (const float*)d_in[1];
    const float* m1_0 = (const float*)d_in[2];
    const float* m2_0 = (const float*)d_in[3];
    const float* mo_0 = (const float*)d_in[4];
    const float* mm_0 = (const float*)d_in[5];
    const float* W1   = (const float*)d_in[6];
    const float* b1   = (const float*)d_in[7];
    const float* W2   = (const float*)d_in[8];
    const float* b2   = (const float*)d_in[9];
    const float* Wo   = (const float*)d_in[10];
    const float* bo   = (const float*)d_in[11];
    const float* Wm   = (const float*)d_in[12];
    const float* bm   = (const float*)d_in[13];
    float* out = (float*)d_out;

    char* ws = (char*)d_ws;
    size_t off = 0;
    auto alloc = [&](size_t bytes) -> void* {
        void* p = ws + off;
        off = (off + bytes + 255) & ~(size_t)255;
        return p;
    };
    _Float16* W1in16  = (_Float16*)alloc((size_t)H_ * D_IN_P * 2);
    _Float16* W1rec16 = (_Float16*)alloc((size_t)H_ * H_ * 2);
    _Float16* W2_16   = (_Float16*)alloc((size_t)H_ * H_ * 2);
    _Float16* Wo16    = (_Float16*)alloc((size_t)D_OUT_P * H_ * 2);
    _Float16* Wm16    = (_Float16*)alloc((size_t)D_OUT_P * H_ * 2);
    _Float16* inp16   = (_Float16*)alloc((size_t)B_ * D_IN_P * 2);
    float*    cur1    = (float*)alloc((size_t)B_ * H_ * 4);
    float*    m1      = (float*)alloc((size_t)B_ * H_ * 4);
    float*    m2      = (float*)alloc((size_t)B_ * H_ * 4);
    float*    mo      = (float*)alloc((size_t)B_ * D_OUT_P * 4);
    float*    mm      = (float*)alloc((size_t)B_ * D_OUT_P * 4);
    _Float16* s1a     = (_Float16*)alloc((size_t)B_ * H_ * 2);
    _Float16* s1b     = (_Float16*)alloc((size_t)B_ * H_ * 2);
    _Float16* s2      = (_Float16*)alloc((size_t)B_ * H_ * 2);
    int*      bar     = (int*)alloc(512);
    (void)ws_size; (void)in_sizes; (void)n_in; (void)out_size;

    const int TPB = 256;
    auto blks = [](int n, int t) { return (n + t - 1) / t; };

    slice_pad_f16_kernel<<<blks(H_ * D_IN_P, TPB), TPB, 0, stream>>>(
        W1, W1in16, H_, H_, D_IN + H_, 0, D_IN, D_IN_P);
    slice_pad_f16_kernel<<<blks(H_ * H_, TPB), TPB, 0, stream>>>(
        W1, W1rec16, H_, H_, D_IN + H_, D_IN, H_, H_);
    slice_pad_f16_kernel<<<blks(H_ * H_, TPB), TPB, 0, stream>>>(
        W2, W2_16, H_, H_, H_, 0, H_, H_);
    slice_pad_f16_kernel<<<blks(D_OUT_P * H_, TPB), TPB, 0, stream>>>(
        Wo, Wo16, D_OUT_P, D_OUT, H_, 0, H_, H_);
    slice_pad_f16_kernel<<<blks(D_OUT_P * H_, TPB), TPB, 0, stream>>>(
        Wm, Wm16, D_OUT_P, D_OUT, H_, 0, H_, H_);
    slice_pad_f16_kernel<<<blks(B_ * D_IN_P, TPB), TPB, 0, stream>>>(
        inp, inp16, B_, B_, D_IN, 0, D_IN, D_IN_P);
    init_state_kernel<<<blks(B_ * H_, TPB), TPB, 0, stream>>>(
        s1_0, m1_0, m2_0, mo_0, mm_0, s1a, m1, m2, mo, mm, bar);

    base_kernel<<<64, 128, 0, stream>>>(inp16, W1in16, b1, cur1);

    // single persistent launch for all 100 timesteps
    snn_persistent<<<NBLOCKS, 128, 0, stream>>>(
        W1rec16, W2_16, Wo16, Wm16, b2, bo, bm, cur1,
        m1, m2, mo, mm, s1a, s1b, s2, out, bar);
}
